// Attn_user_47863115547245
// MI455X (gfx1250) — matrix operand order repaired and checked by reading, not compiled, not run on hardware
//
#include <hip/hip_runtime.h>
#include <hip/hip_bf16.h>
#include <math.h>

#define STATE_LEN 2048
#define SEQ_LEN   4096
#define HIDDEN    1024

typedef __attribute__((ext_vector_type(16))) __bf16 v16bf;
typedef __attribute__((ext_vector_type(8)))  float  v8f;

// ---------------- f32 -> bf16 (round-to-nearest-even) ----------------
__device__ __forceinline__ unsigned short f2bf(float x) {
    unsigned int u = __float_as_uint(x);
    u += 0x7FFFu + ((u >> 16) & 1u);
    return (unsigned short)(u >> 16);
}

__global__ void cvt_bf16_kernel(const float* __restrict__ src,
                                unsigned short* __restrict__ dst, int n) {
    int i = blockIdx.x * blockDim.x + threadIdx.x;
    int stride = gridDim.x * blockDim.x;
    for (; i < n; i += stride) dst[i] = f2bf(src[i]);
}

// comb = bf16(w1*user + w2*social)
__global__ void comb_bf16_kernel(const float* __restrict__ user,
                                 const float* __restrict__ social,
                                 const float* __restrict__ w1p,
                                 const float* __restrict__ w2p,
                                 unsigned short* __restrict__ dst, int n) {
    float w1 = w1p[0], w2 = w2p[0];
    int i = blockIdx.x * blockDim.x + threadIdx.x;
    int stride = gridDim.x * blockDim.x;
    for (; i < n; i += stride) dst[i] = f2bf(w1 * user[i] + w2 * social[i]);
}

// ---------------- WMMA fragment loaders (contiguous k = 16h + s, same for A and B) ----------------
// A fragment: 16x32 bf16, source row-major [M, lda].
// lanes 0-15 hold row m=lane, K = {kb..kb+15}
// lanes 16-31 hold row m=lane-16, K = {kb+16..kb+31}
__device__ __forceinline__ v16bf load_frag_a(const unsigned short* __restrict__ A,
                                             int lda, int mbase, int kb) {
    int lane = threadIdx.x & 31;
    int m = lane & 15;
    int half = lane >> 4;
    const unsigned short* p = A + (size_t)(mbase + m) * lda + kb + half * 16;
    union { uint4 u; __bf16 h[8]; } c0, c1;
    c0.u = *reinterpret_cast<const uint4*>(p);
    c1.u = *reinterpret_cast<const uint4*>(p + 8);
    v16bf f;
#pragma unroll
    for (int i = 0; i < 8; ++i) { f[i] = c0.h[i]; f[8 + i] = c1.h[i]; }
    return f;
}

// B fragment: 32x16 bf16, loaded from Bt = row-major [N, ldb] (i.e. B transposed).
// lane holds column n=lane&15, 16 consecutive K starting at kb + (lane>>4)*16.
__device__ __forceinline__ v16bf load_frag_b(const unsigned short* __restrict__ Bt,
                                             int ldb, int nbase, int kb) {
    int lane = threadIdx.x & 31;
    int n = lane & 15;
    int half = lane >> 4;
    const unsigned short* p = Bt + (size_t)(nbase + n) * ldb + kb + half * 16;
    union { uint4 u[2]; v16bf v; } c;
    c.u[0] = *reinterpret_cast<const uint4*>(p);
    c.u[1] = *reinterpret_cast<const uint4*>(p + 8);
    return c.v;
}

// ---------------- GEMM: C[M,N] = A[M,K] @ Bt[N,K]^T (+bias[n]) ----------------
// Block = 256 threads = 8 waves arranged 2(M) x 4(N).
// Each wave computes a 32(M) x 64(N) tile: 2 A frags + 4 B frags -> 8 WMMAs/k-step.
// Block tile = 64(M) x 256(N). Grid: x = N/256, y = M/64.
// Output: either f32 (C_f32) or bf16 (C_bf16), row-major [M, N].
__global__ void __launch_bounds__(256)
wmma_gemm_bt_kernel(const unsigned short* __restrict__ A,
                    const unsigned short* __restrict__ Bt,
                    const float* __restrict__ bias,
                    float* __restrict__ C_f32,
                    unsigned short* __restrict__ C_bf16,
                    int M, int N, int K) {
    int wave  = threadIdx.x >> 5;          // 0..7
    int mtile = wave >> 2;                 // 0..1
    int ntile = wave & 3;                  // 0..3
    int mbase = blockIdx.y * 64 + mtile * 32;
    int nbase = blockIdx.x * 256 + ntile * 64;
    if (mbase >= M || nbase >= N) return;

    v8f acc[2][4];
#pragma unroll
    for (int i = 0; i < 2; ++i)
#pragma unroll
        for (int t = 0; t < 4; ++t) acc[i][t] = (v8f){};

    for (int kb = 0; kb < K; kb += 32) {
        v16bf a0 = load_frag_a(A, K, mbase,      kb);
        v16bf a1 = load_frag_a(A, K, mbase + 16, kb);
#pragma unroll
        for (int t = 0; t < 4; ++t) {
            v16bf b = load_frag_b(Bt, K, nbase + t * 16, kb);
            acc[0][t] = __builtin_amdgcn_wmma_f32_16x16x32_bf16(
                false, a0, false, b, (short)0, acc[0][t], false, false);
            acc[1][t] = __builtin_amdgcn_wmma_f32_16x16x32_bf16(
                false, a1, false, b, (short)0, acc[1][t], false, false);
        }
    }

    // D layout: element r -> row M = mbase + i*16 + (lane>>4)*8 + r,
    //           col N = nbase + t*16 + (lane&15)
    int lane = threadIdx.x & 31;
    int ncol = lane & 15;
    int rowoff = (lane >> 4) * 8;
#pragma unroll
    for (int t = 0; t < 4; ++t) {
        int col = nbase + t * 16 + ncol;
        float bv = bias ? bias[col] : 0.0f;
#pragma unroll
        for (int i = 0; i < 2; ++i) {
#pragma unroll
            for (int r = 0; r < 8; ++r) {
                int row = mbase + i * 16 + rowoff + r;
                float v = acc[i][t][r] + bv;
                size_t idx = (size_t)row * N + col;
                if (C_f32) C_f32[idx] = v;
                else       C_bf16[idx] = f2bf(v);
            }
        }
    }
}

// ---------------- in-place row softmax over `cols` ----------------
__global__ void __launch_bounds__(256)
softmax_rows_kernel(float* __restrict__ x, int cols) {
    __shared__ float red[256];
    float* p = x + (size_t)blockIdx.x * cols;
    int tid = threadIdx.x;

    float m = -INFINITY;
    for (int i = tid; i < cols; i += 256) m = fmaxf(m, p[i]);
    red[tid] = m; __syncthreads();
    for (int s = 128; s > 0; s >>= 1) {
        if (tid < s) red[tid] = fmaxf(red[tid], red[tid + s]);
        __syncthreads();
    }
    m = red[0]; __syncthreads();

    float sum = 0.0f;
    for (int i = tid; i < cols; i += 256) sum += __expf(p[i] - m);
    red[tid] = sum; __syncthreads();
    for (int s = 128; s > 0; s >>= 1) {
        if (tid < s) red[tid] += red[tid + s];
        __syncthreads();
    }
    float inv = 1.0f / red[0];

    for (int i = tid; i < cols; i += 256) p[i] = __expf(p[i] - m) * inv;
}

// ---------------- launch ----------------
extern "C" void kernel_launch(void* const* d_in, const int* in_sizes, int n_in,
                              void* d_out, int out_size, void* d_ws, size_t ws_size,
                              hipStream_t stream) {
    const float* user = (const float*)d_in[0];   // [2048,1024]
    const float* id   = (const float*)d_in[1];   // [4096,1024]
    const float* soc  = (const float*)d_in[2];   // [2048,1024]
    const float* W    = (const float*)d_in[3];   // [1024,1024]
    const float* b    = (const float*)d_in[4];   // [1024]
    const float* w1   = (const float*)d_in[5];
    const float* w2   = (const float*)d_in[6];
    float* out = (float*)d_out;                  // [2048,4096]

    // workspace layout (bytes)
    char* ws = (char*)d_ws;
    unsigned short* id_bf   = (unsigned short*)(ws);                       // 8 MB
    unsigned short* W_bf    = (unsigned short*)(ws + (8u  << 20));         // 2 MB
    unsigned short* comb_bf = (unsigned short*)(ws + (10u << 20));         // 4 MB
    unsigned short* proj_bf = (unsigned short*)(ws + (14u << 20));         // 8 MB

    // ---- prep: f32 -> bf16 (and weighted combine) ----
    {
        int n_id = SEQ_LEN * HIDDEN;
        cvt_bf16_kernel<<<dim3(512), dim3(256), 0, stream>>>(id, id_bf, n_id);
        int n_w = HIDDEN * HIDDEN;
        cvt_bf16_kernel<<<dim3(256), dim3(256), 0, stream>>>(W, W_bf, n_w);
        int n_c = STATE_LEN * HIDDEN;
        comb_bf16_kernel<<<dim3(512), dim3(256), 0, stream>>>(user, soc, w1, w2, comb_bf, n_c);
    }

    // ---- GEMM1: proj[q,h] = id[q,:] . W[h,:] + b[h]  (M=4096, N=1024, K=1024) ----
    {
        dim3 grid(HIDDEN / 256, SEQ_LEN / 64);
        wmma_gemm_bt_kernel<<<grid, dim3(256), 0, stream>>>(
            id_bf, W_bf, b, nullptr, proj_bf, SEQ_LEN, HIDDEN, HIDDEN);
    }

    // ---- GEMM2: energies[s,q] = comb[s,:] . proj[q,:]  (M=2048, N=4096, K=1024) ----
    {
        dim3 grid(SEQ_LEN / 256, STATE_LEN / 64);
        wmma_gemm_bt_kernel<<<grid, dim3(256), 0, stream>>>(
            comb_bf, proj_bf, nullptr, out, nullptr, STATE_LEN, SEQ_LEN, HIDDEN);
    }

    // ---- softmax over seq axis, in place on d_out ----
    softmax_rows_kernel<<<dim3(STATE_LEN), dim3(256), 0, stream>>>(out, SEQ_LEN);
}
